// TPMCAM_29978871726248
// MI455X (gfx1250) — matrix operand order repaired and checked by reading, not compiled, not run on hardware
//
#include <hip/hip_runtime.h>
#include <hip/hip_bf16.h>
#include <math.h>
#include <stdint.h>

// ---------------- problem constants ----------------
constexpr int  Nn   = 128000;     // nodes
constexpr int  Gg   = 64;         // graphs
constexpr int  Ee   = 1024000;    // GIN edges
constexpr int  Epc  = 512000;     // pc / mc edges (before self loops)
constexpr int  NPG  = Nn / Gg;    // 2000 nodes per graph
constexpr int  Ll   = 4;

typedef __attribute__((ext_vector_type(16))) __bf16 v16bf;
typedef __attribute__((ext_vector_type(8)))  float  v8f;

// ---------------- small helpers ----------------
__device__ inline void atomicMaxF(float* addr, float v) {
  // sign-split trick: works for IEEE-754 ordering with -inf init
  if (v >= 0.0f) atomicMax((int*)addr, __float_as_int(v));
  else           atomicMin((unsigned int*)addr, __float_as_uint(v));
}

__device__ inline void edge_sd(const int* __restrict__ src, const int* __restrict__ dst,
                               int E, int e, int& s, int& d) {
  if (e < E) { s = src[e]; d = dst[e]; } else { s = e - E; d = s; }  // self loops appended
}

__device__ inline float elu1(float x) { return x > 0.0f ? x : (__expf(x) - 1.0f); }

// ---------------- generic fills / accumulate ----------------
__global__ void fill_f32(float* __restrict__ p, float v, int n) {
  int i = blockIdx.x * 256 + threadIdx.x;
  if (i < n) p[i] = v;
}

__global__ void accum_f32(float* __restrict__ dst, const float* __restrict__ src, int n) {
  int i = blockIdx.x * 256 + threadIdx.x;
  if (i < n) dst[i] += src[i];
}

// ---------------- WMMA GEMM: out[N,128] = A[N,128] @ W[128,128] (+bias) ----------------
// block = 256 threads (8 waves); each block does 128 rows; wave w -> rows [base+16w, +16)
// W (fp32, row-major k x n) is DMA'd to LDS via global_load_async_to_lds_b128 in 16KB
// chunks, then converted/transposed to bf16: Wt[n*128 + k].
__global__ __launch_bounds__(256) void gemm128_wmma_bf16(
    const float* __restrict__ A, const float* __restrict__ Wg,
    const float* __restrict__ bias, float* __restrict__ out) {
  __shared__ float  Wstage[4096];                    // 16 KB async staging
  __shared__ __bf16 Wt[128 * 128];                   // 32 KB bf16, transposed
  const int tid = threadIdx.x;

  #pragma unroll 1
  for (int cch = 0; cch < 4; ++cch) {                // 4 chunks of 32 rows of W
    const float* gsrc = Wg + cch * 4096;
    #pragma unroll
    for (int it = 0; it < 4; ++it) {                 // 256 thr x 16B x 4 = 16 KB
      const int idx = (it * 256 + tid) * 4;          // float index, 16B granules
      uint64_t ga = (uint64_t)(uintptr_t)(gsrc + idx);
      uint32_t la = (uint32_t)(uintptr_t)(&Wstage[idx]);   // LDS byte offset (addr[31:0])
      asm volatile("global_load_async_to_lds_b128 %0, %1, off"
                   :: "v"(la), "v"(ga) : "memory");
    }
    asm volatile("s_wait_asynccnt 0" ::: "memory");  // ASYNCcnt == 0 for this wave
    __syncthreads();
    for (int i = tid; i < 4096; i += 256) {
      const int gi = cch * 4096 + i;
      const int k = gi >> 7, n = gi & 127;
      Wt[n * 128 + k] = (__bf16)Wstage[i];
    }
    __syncthreads();                                 // protect Wstage reuse
  }

  const int wave = tid >> 5;
  const int lane = tid & 31;
  const int half = lane >> 4;                        // 0: lanes 0-15, 1: lanes 16-31
  const int lrow = lane & 15;
  const int rowBase = blockIdx.x * 128 + wave * 16;  // Nn = 1000*128, exact

  v8f acc[8];
  #pragma unroll
  for (int t = 0; t < 8; ++t) acc[t] = (v8f){};

  const float* ap = A + (size_t)(rowBase + lrow) * 128;

  #pragma unroll
  for (int kc = 0; kc < 4; ++kc) {                   // K = 128 in 4 chunks of 32
    const int kbase = kc * 32;
    // A fragment: lane half selects K 0-15 / 16-31 (contiguous, matches B)
    v16bf a;
    #pragma unroll
    for (int j = 0; j < 16; ++j) a[j] = (__bf16)ap[kbase + half * 16 + j];

    #pragma unroll
    for (int nt = 0; nt < 8; ++nt) {                 // 8 column tiles of 16
      // B fragment: lane -> column n, halves split K 0-15 / 16-31; contiguous in Wt
      const __bf16* bp = &Wt[(nt * 16 + lrow) * 128 + kbase + half * 16];
      v16bf b;
      #pragma unroll
      for (int j = 0; j < 16; ++j) b[j] = bp[j];
      acc[nt] = __builtin_amdgcn_wmma_f32_16x16x32_bf16(
          false, a, false, b, (short)0, acc[nt], false, false);
    }
  }

  // C layout: vgpr r -> M = r + 8*half ; lane lrow -> N
  #pragma unroll
  for (int nt = 0; nt < 8; ++nt) {
    const int n = nt * 16 + lrow;
    const float bv = bias ? bias[n] : 0.0f;
    #pragma unroll
    for (int r = 0; r < 8; ++r) {
      const int m = rowBase + half * 8 + r;
      out[(size_t)m * 128 + n] = acc[nt][r] + bv;
    }
  }
}

// ---------------- in_dim=3 GEMM (tiny K, plain VALU) ----------------
__global__ void gemm_in3(const float* __restrict__ t3, const float* __restrict__ W,
                         const float* __restrict__ b, float* __restrict__ out) {
  int i = blockIdx.x * 256 + threadIdx.x;            // N*128 threads
  int n = i >> 7, f = i & 127;
  const float* tr = t3 + n * 3;
  float v = tr[0] * W[f] + tr[1] * W[128 + f] + tr[2] * W[256 + f];
  out[i] = b ? v + b[f] : v;
}

// ---------------- segment-mean aggregation ----------------
__global__ void edge_accum3(const float* __restrict__ x, const int* __restrict__ src,
                            const int* __restrict__ dst, int nE,
                            float* __restrict__ acc, float* __restrict__ cnt) {
  int e = blockIdx.x * 256 + threadIdx.x;
  if (e >= nE) return;
  int s = src[e], d = dst[e];
  atomicAdd(acc + d * 3 + 0, x[s * 3 + 0]);
  atomicAdd(acc + d * 3 + 1, x[s * 3 + 1]);
  atomicAdd(acc + d * 3 + 2, x[s * 3 + 2]);
  atomicAdd(cnt + d, 1.0f);
}

__global__ void edge_accum128(const float* __restrict__ x, const int* __restrict__ src,
                              const int* __restrict__ dst, int nE,
                              float* __restrict__ acc, float* __restrict__ cnt) {
  int gid = blockIdx.x * 256 + threadIdx.x;          // nE*32 threads
  int e = gid >> 5;
  if (e >= nE) return;
  int c = gid & 31;
  int s = src[e], d = dst[e];
  const float4 v = ((const float4*)(x + (size_t)s * 128))[c];
  float* o = acc + (size_t)d * 128 + c * 4;
  atomicAdd(o + 0, v.x); atomicAdd(o + 1, v.y);
  atomicAdd(o + 2, v.z); atomicAdd(o + 3, v.w);
  if (c == 0) atomicAdd(cnt + d, 1.0f);
}

__global__ void gin_pre3(const float* __restrict__ x, float* __restrict__ acc,
                         const float* __restrict__ cnt) {
  int i = blockIdx.x * 256 + threadIdx.x;            // N*3
  if (i >= Nn * 3) return;
  int r = i / 3;
  acc[i] = x[i] + acc[i] / fmaxf(cnt[r], 1.0f);      // t = x + mean_agg
}

__global__ void gin_pre128(const float* __restrict__ h, float* __restrict__ acc,
                           const float* __restrict__ cnt) {
  int i = blockIdx.x * 256 + threadIdx.x;            // N*128
  int r = i >> 7;
  acc[i] = h[i] + acc[i] / fmaxf(cnt[r], 1.0f);
}

// ---------------- BatchNorm (batch stats) + ReLU ----------------
__global__ void bn_stats(const float* __restrict__ z, float* __restrict__ sums) {
  int f = threadIdx.x;                               // 128 threads, 250 blocks x 512 rows
  const float* p = z + (size_t)blockIdx.x * 512 * 128 + f;
  float s = 0.0f, q = 0.0f;
  for (int r = 0; r < 512; ++r) { float v = p[(size_t)r * 128]; s += v; q += v * v; }
  atomicAdd(sums + f, s);
  atomicAdd(sums + 128 + f, q);
}

__global__ void bn_coef(const float* __restrict__ sums, const float* __restrict__ g,
                        const float* __restrict__ beta, float* __restrict__ coef) {
  int f = threadIdx.x;
  float mu  = sums[f] * (1.0f / (float)Nn);
  float var = sums[128 + f] * (1.0f / (float)Nn) - mu * mu;
  float sc  = g[f] * rsqrtf(var + 1e-5f);
  coef[f] = sc;
  coef[128 + f] = beta[f] - mu * sc;
}

__global__ void bn_relu(float* __restrict__ z, const float* __restrict__ coef) {
  int i = blockIdx.x * 256 + threadIdx.x;
  int f = i & 127;
  z[i] = fmaxf(z[i] * coef[f] + coef[128 + f], 0.0f);
}

// ---------------- GAT pieces ----------------
__global__ void gat_scores(const float* __restrict__ h, const float* __restrict__ asrc,
                           const float* __restrict__ adst,
                           float* __restrict__ ssrc, float* __restrict__ sdst) {
  int n = blockIdx.x * 256 + threadIdx.x;
  if (n >= Nn) return;
  const float* p = h + (size_t)n * 128;
  float a = 0.0f, b = 0.0f;
  #pragma unroll 4
  for (int j = 0; j < 128; ++j) { float v = p[j]; a += v * asrc[j]; b += v * adst[j]; }
  ssrc[n] = a; sdst[n] = b;
}

__global__ void gat_edge_max(const int* __restrict__ src, const int* __restrict__ dst,
                             int E, int nE, const float* __restrict__ ssrc,
                             const float* __restrict__ sdst, float* __restrict__ mmax) {
  int e = blockIdx.x * 256 + threadIdx.x;
  if (e >= nE) return;
  int s, d; edge_sd(src, dst, E, e, s, d);
  float v = ssrc[s] + sdst[d];
  v = v > 0.0f ? v : 0.2f * v;                       // leaky_relu(0.2)
  atomicMaxF(mmax + d, v);
}

__global__ void gat_edge_exp(const int* __restrict__ src, const int* __restrict__ dst,
                             int E, int nE, const float* __restrict__ ssrc,
                             const float* __restrict__ sdst, const float* __restrict__ mmax,
                             float* __restrict__ alpha, float* __restrict__ ssum) {
  int e = blockIdx.x * 256 + threadIdx.x;
  if (e >= nE) return;
  int s, d; edge_sd(src, dst, E, e, s, d);
  float v = ssrc[s] + sdst[d];
  v = v > 0.0f ? v : 0.2f * v;
  float a = __expf(v - mmax[d]);
  alpha[e] = a;
  atomicAdd(ssum + d, a);
}

__global__ void gat_edge_aggr(const int* __restrict__ src, const int* __restrict__ dst,
                              int E, int nE, const float* __restrict__ h,
                              const float* __restrict__ alpha, const float* __restrict__ ssum,
                              float* __restrict__ acc) {
  int gid = blockIdx.x * 256 + threadIdx.x;          // nE*32 threads
  int e = gid >> 5;
  if (e >= nE) return;
  int c = gid & 31;
  int s, d; edge_sd(src, dst, E, e, s, d);
  float coef = alpha[e] / (ssum[d] + 1e-16f);
  const float4 v = ((const float4*)(h + (size_t)s * 128))[c];
  float* o = acc + (size_t)d * 128 + c * 4;
  atomicAdd(o + 0, v.x * coef); atomicAdd(o + 1, v.y * coef);
  atomicAdd(o + 2, v.z * coef); atomicAdd(o + 3, v.w * coef);
}

__global__ void gat_finish0(const float* __restrict__ acc, const float* __restrict__ bias,
                            float* __restrict__ hpc) {
  int i = blockIdx.x * 256 + threadIdx.x;
  int f = i & 127;
  hpc[i] = elu1(acc[i] + bias[f]);
}

__global__ void gat_finish1(const float* __restrict__ acc, const float* __restrict__ bias,
                            const float* __restrict__ hpc, float* __restrict__ hd) {
  int i = blockIdx.x * 256 + threadIdx.x;
  int f = i & 127;
  hd[i] = 0.5f * (hpc[i] + elu1(acc[i] + bias[f]));
}

// ---------------- pooling / output ----------------
__global__ void graph_mean_accum(const float* __restrict__ h, float* __restrict__ gp) {
  int g = blockIdx.x, f = threadIdx.x;               // Gg blocks x 128
  const float* p = h + (size_t)g * NPG * 128 + f;
  float s = 0.0f;
  for (int r = 0; r < NPG; ++r) s += p[(size_t)r * 128];
  gp[g * 128 + f] += s * (1.0f / (float)NPG);
}

__global__ void write_node_embed(const float* __restrict__ np, const float* __restrict__ hd,
                                 float* __restrict__ out) {
  int i = blockIdx.x * 256 + threadIdx.x;            // N*128
  int row = i >> 7, f = i & 127;
  out[(size_t)row * 256 + f]       = np[i];
  out[(size_t)row * 256 + 128 + f] = hd[i];
}

__global__ void write_graph_embed(const float* __restrict__ gp, const float* __restrict__ hd,
                                  float* __restrict__ out) {                 // out at N*256
  int g = blockIdx.x, f = threadIdx.x;
  const float* p = hd + (size_t)g * NPG * 128 + f;
  float s = 0.0f;
  for (int r = 0; r < NPG; ++r) s += p[(size_t)r * 128];
  out[(size_t)g * 256 + f]       = gp[g * 128 + f];
  out[(size_t)g * 256 + 128 + f] = s * (1.0f / (float)NPG);
}

// ---------------- host orchestration ----------------
extern "C" void kernel_launch(void* const* d_in, const int* in_sizes, int n_in,
                              void* d_out, int out_size, void* d_ws, size_t ws_size,
                              hipStream_t stream) {
  (void)in_sizes; (void)n_in; (void)out_size; (void)ws_size;
  const float* x        = (const float*)d_in[0];
  const int*   ei       = (const int*)d_in[1];
  const int*   pc       = (const int*)d_in[2];
  const int*   mc       = (const int*)d_in[3];
  const float* gin0_W1  = (const float*)d_in[6];
  const float* gin0_b1  = (const float*)d_in[7];
  const float* gin0_g   = (const float*)d_in[8];
  const float* gin0_be  = (const float*)d_in[9];
  const float* gin0_W2  = (const float*)d_in[10];
  const float* gin0_b2  = (const float*)d_in[11];
  const float* ginr_W1  = (const float*)d_in[12];
  const float* ginr_b1  = (const float*)d_in[13];
  const float* ginr_g   = (const float*)d_in[14];
  const float* ginr_be  = (const float*)d_in[15];
  const float* ginr_W2  = (const float*)d_in[16];
  const float* ginr_b2  = (const float*)d_in[17];
  const float* gat0_W   = (const float*)d_in[18];
  const float* gat0_as  = (const float*)d_in[19];
  const float* gat0_ad  = (const float*)d_in[20];
  const float* gat0_b   = (const float*)d_in[21];
  const float* gatr_W   = (const float*)d_in[22];
  const float* gatr_as  = (const float*)d_in[23];
  const float* gatr_ad  = (const float*)d_in[24];
  const float* gatr_b   = (const float*)d_in[25];
  float* out = (float*)d_out;

  const int* ei_s = ei;       const int* ei_d = ei + Ee;
  const int* pc_s = pc;       const int* pc_d = pc + Epc;
  const int* mc_s = mc;       const int* mc_d = mc + Epc;

  // ---- workspace layout (floats): 5 big [N,128] + small scratch ----
  const size_t NH = (size_t)Nn * 128;
  float* ws     = (float*)d_ws;
  float* b_h    = ws;                ws += NH;   // gin h / gat hpc
  float* b_agg  = ws;                ws += NH;   // agg,t / gat h=hd@W
  float* b_z    = ws;                ws += NH;   // gin z / gat acc
  float* b_np   = ws;                ws += NH;   // node_pool
  float* b_hd   = ws;                ws += NH;   // dghan hidden
  float* cnt    = ws;                ws += Nn;
  float* ssrc   = ws;                ws += Nn;
  float* sdst   = ws;                ws += Nn;
  float* mmax   = ws;                ws += Nn;
  float* ssum   = ws;                ws += Nn;
  float* alpha  = ws;                ws += (Epc + Nn);
  float* acc3   = ws;                ws += (size_t)Nn * 3;
  float* sums   = ws;                ws += 256;
  float* coef   = ws;                ws += 256;
  float* gpool  = ws;                ws += (size_t)Gg * 128;

  const int EB   = (int)(NH / 256);                  // 64000 blocks for N*128 elementwise
  const int GEMB = Nn / 128;                         // 1000 blocks for WMMA GEMM
  const int nEg  = Epc + Nn;                         // gat edges incl. self loops (640000)

  // ================= GIN branch =================
  hipMemsetAsync(b_np,  0, NH * 4, stream);
  hipMemsetAsync(gpool, 0, (size_t)Gg * 128 * 4, stream);

  // layer 0 (in_dim = 3)
  hipMemsetAsync(cnt,  0, (size_t)Nn * 4, stream);
  hipMemsetAsync(acc3, 0, (size_t)Nn * 3 * 4, stream);
  edge_accum3<<<Ee / 256, 256, 0, stream>>>(x, ei_s, ei_d, Ee, acc3, cnt);
  gin_pre3<<<(Nn * 3) / 256, 256, 0, stream>>>(x, acc3, cnt);
  gemm_in3<<<EB, 256, 0, stream>>>(acc3, gin0_W1, gin0_b1, b_z);
  hipMemsetAsync(sums, 0, 256 * 4, stream);
  bn_stats<<<Nn / 512, 128, 0, stream>>>(b_z, sums);
  bn_coef<<<1, 128, 0, stream>>>(sums, gin0_g, gin0_be, coef);
  bn_relu<<<EB, 256, 0, stream>>>(b_z, coef);
  gemm128_wmma_bf16<<<GEMB, 256, 0, stream>>>(b_z, gin0_W2, gin0_b2, b_h);
  accum_f32<<<EB, 256, 0, stream>>>(b_np, b_h, (int)NH);
  graph_mean_accum<<<Gg, 128, 0, stream>>>(b_h, gpool);

  // layers 1..3
  for (int l = 0; l < Ll - 1; ++l) {
    hipMemsetAsync(cnt,   0, (size_t)Nn * 4, stream);
    hipMemsetAsync(b_agg, 0, NH * 4, stream);
    edge_accum128<<<(Ee * 32) / 256, 256, 0, stream>>>(b_h, ei_s, ei_d, Ee, b_agg, cnt);
    gin_pre128<<<EB, 256, 0, stream>>>(b_h, b_agg, cnt);
    gemm128_wmma_bf16<<<GEMB, 256, 0, stream>>>(b_agg, ginr_W1 + (size_t)l * 16384,
                                                ginr_b1 + l * 128, b_z);
    hipMemsetAsync(sums, 0, 256 * 4, stream);
    bn_stats<<<Nn / 512, 128, 0, stream>>>(b_z, sums);
    bn_coef<<<1, 128, 0, stream>>>(sums, ginr_g + l * 128, ginr_be + l * 128, coef);
    bn_relu<<<EB, 256, 0, stream>>>(b_z, coef);
    gemm128_wmma_bf16<<<GEMB, 256, 0, stream>>>(b_z, ginr_W2 + (size_t)l * 16384,
                                                ginr_b2 + l * 128, b_h);
    accum_f32<<<EB, 256, 0, stream>>>(b_np, b_h, (int)NH);
    graph_mean_accum<<<Gg, 128, 0, stream>>>(b_h, gpool);
  }

  // ================= DGHAN (GAT) branch =================
  for (int l = 0; l < Ll; ++l) {
    for (int c = 0; c < 2; ++c) {
      const int* es = c == 0 ? pc_s : mc_s;
      const int* ed = c == 0 ? pc_d : mc_d;
      // h = input @ W  (-> b_agg)
      if (l == 0) {
        gemm_in3<<<EB, 256, 0, stream>>>(x, gat0_W + (size_t)c * 384, nullptr, b_agg);
      } else {
        gemm128_wmma_bf16<<<GEMB, 256, 0, stream>>>(
            b_hd, gatr_W + (size_t)((l - 1) * 2 + c) * 16384, nullptr, b_agg);
      }
      const float* as = l == 0 ? gat0_as + c * 128 : gatr_as + ((l - 1) * 2 + c) * 128;
      const float* ad = l == 0 ? gat0_ad + c * 128 : gatr_ad + ((l - 1) * 2 + c) * 128;
      const float* bb = l == 0 ? gat0_b  + c * 128 : gatr_b  + ((l - 1) * 2 + c) * 128;

      gat_scores<<<Nn / 256, 256, 0, stream>>>(b_agg, as, ad, ssrc, sdst);
      fill_f32<<<Nn / 256, 256, 0, stream>>>(mmax, -INFINITY, Nn);
      hipMemsetAsync(ssum, 0, (size_t)Nn * 4, stream);
      hipMemsetAsync(b_z,  0, NH * 4, stream);
      gat_edge_max<<<nEg / 256, 256, 0, stream>>>(es, ed, Epc, nEg, ssrc, sdst, mmax);
      gat_edge_exp<<<nEg / 256, 256, 0, stream>>>(es, ed, Epc, nEg, ssrc, sdst, mmax,
                                                  alpha, ssum);
      gat_edge_aggr<<<(nEg * 32) / 256, 256, 0, stream>>>(es, ed, Epc, nEg, b_agg,
                                                          alpha, ssum, b_z);
      if (c == 0) gat_finish0<<<EB, 256, 0, stream>>>(b_z, bb, b_h);
      else        gat_finish1<<<EB, 256, 0, stream>>>(b_z, bb, b_h, b_hd);
    }
  }

  // ================= fuse / outputs =================
  write_node_embed<<<EB, 256, 0, stream>>>(b_np, b_hd, out);
  write_graph_embed<<<Gg, 128, 0, stream>>>(gpool, b_hd, out + (size_t)Nn * 256);
}
